// CausalSelfAttention_24464133718700
// MI455X (gfx1250) — matrix operand order repaired and checked by reading, not compiled, not run on hardware
//
#include <hip/hip_runtime.h>
#include <stdint.h>

// ---------------------------------------------------------------------------
// CDNA5 (gfx1250) causal self-attention, bf16 WMMA with f32 accumulation.
// GEMM tiles staged via GLOBAL_LOAD_ASYNC_TO_LDS (ASYNCcnt); flash-attention
// softmax reductions done with DPP16 butterflies (no LDS round-trips).
// ---------------------------------------------------------------------------

typedef unsigned short u16;
typedef __attribute__((ext_vector_type(16))) __bf16 v16bf;
typedef __attribute__((ext_vector_type(16))) u16    v16u;
typedef __attribute__((ext_vector_type(8)))  u16    v8u;
typedef __attribute__((ext_vector_type(8)))  float  v8f;

__device__ __forceinline__ u16 f2bf(float f) {
  unsigned int u = __float_as_uint(f);
  u += 0x7FFFu + ((u >> 16) & 1u);          // round-to-nearest-even
  return (u16)(u >> 16);
}
__device__ __forceinline__ float bf2f(u16 h) {
  return __uint_as_float(((unsigned int)h) << 16);
}
__device__ __forceinline__ v16u cat16(v8u a, v8u b) {
  v16u r;
#pragma unroll
  for (int i = 0; i < 8; ++i) { r[i] = a[i]; r[i + 8] = b[i]; }
  return r;
}
// D = A(16x32 bf16) x B(32x16 bf16) + C(16x16 f32)
__device__ __forceinline__ v8f wmma_bf16(v16u a, v16u b, v8f c) {
  return __builtin_amdgcn_wmma_f32_16x16x32_bf16(
      false, __builtin_bit_cast(v16bf, a),
      false, __builtin_bit_cast(v16bf, b),
      (short)0, c, false, false);
}

// ---- DPP16 16-lane reductions (VALU-only, co-executable, no DS waits) -----
// quad_perm(1,0,3,2)=0xB1 (xor1), quad_perm(2,3,0,1)=0x4E (xor2),
// row_half_mirror=0x141 (crosses 4-boundary), row_mirror=0x140 (crosses 8).
template <int CTRL>
__device__ __forceinline__ float dpp_mov(float x) {
  return __int_as_float(__builtin_amdgcn_update_dpp(
      0, __float_as_int(x), CTRL, 0xf, 0xf, true));
}
__device__ __forceinline__ float redmax16(float x) {
  x = fmaxf(x, dpp_mov<0xB1>(x));
  x = fmaxf(x, dpp_mov<0x4E>(x));
  x = fmaxf(x, dpp_mov<0x141>(x));
  x = fmaxf(x, dpp_mov<0x140>(x));
  return x;
}
__device__ __forceinline__ float redsum16(float x) {
  x += dpp_mov<0xB1>(x);
  x += dpp_mov<0x4E>(x);
  x += dpp_mov<0x141>(x);
  x += dpp_mov<0x140>(x);
  return x;
}

// Async global->LDS 32B copy (two B128 transfers; INST_OFFSET applies to both
// the LDS and global addresses per CDNA5 ISA sec 15.18.3).  Low 32 bits of a
// generic LDS pointer are the LDS byte address (flat aperture truncation).
__device__ __forceinline__ void async_copy32B(void* lds, const void* g) {
  unsigned           l = (unsigned)(uintptr_t)lds;
  unsigned long long a = (unsigned long long)(uintptr_t)g;
  asm volatile("global_load_async_to_lds_b128 %0, %1, off\n\t"
               "global_load_async_to_lds_b128 %0, %1, off offset:16"
               :: "v"(l), "v"(a) : "memory");
}
__device__ __forceinline__ void wait_async0() {
  asm volatile("s_wait_asynccnt 0x0" ::: "memory");
}

// ---------------------------------------------------------------------------
// fp32 -> bf16 cast
// ---------------------------------------------------------------------------
__global__ void cvt_bf16_kernel(const float* __restrict__ in,
                                u16* __restrict__ out, int n) {
  int i = blockIdx.x * blockDim.x + threadIdx.x;
  if (i < n) out[i] = f2bf(in[i]);
}

// ---------------------------------------------------------------------------
// C[M,N] = A[M,K] * B[N,K]^T   (both bf16, K-contiguous; async-LDS staged)
// block: 256 thr (8 waves), tile 128x64, wave tile 32x32 (2x2 WMMA), K step 32
// ---------------------------------------------------------------------------
template <typename TOUT>
__global__ void gemm_abt_kernel(const u16* __restrict__ A,
                                const u16* __restrict__ B,
                                TOUT* __restrict__ C, int M, int N, int K) {
  __shared__ u16 As[128][40];   // +8 halves pad, rows stay 16B aligned
  __shared__ u16 Bs[64][40];
  const int t    = threadIdx.x;
  const int lane = t & 31, wid = t >> 5;
  const int wm   = wid >> 1, wn = wid & 1;
  const int m0   = blockIdx.y * 128, n0 = blockIdx.x * 64;
  const int lm   = lane & 15, lh = lane >> 4;

  v8f c[2][2] = {};

  for (int k0 = 0; k0 < K; k0 += 32) {
    { // A tile: 128 rows x 32 halves, 2 threads/row, 32B each, async to LDS
      int row = t >> 1, off = (t & 1) * 16;
      async_copy32B(&As[row][off], A + (size_t)(m0 + row) * K + k0 + off);
    }
    if (t < 128) { // B tile: 64 rows x 32 halves
      int row = t >> 1, off = (t & 1) * 16;
      async_copy32B(&Bs[row][off], B + (size_t)(n0 + row) * K + k0 + off);
    }
    wait_async0();       // this wave's async transfers landed in LDS
    __syncthreads();     // all waves' transfers visible

    v16u a[2], b[2];
#pragma unroll
    for (int mi = 0; mi < 2; ++mi) { // A frag: row = lane%16, K chunks by lane/16
      const u16* r = &As[wm * 32 + mi * 16 + lm][0];
      a[mi] = cat16(*(const v8u*)&r[lh * 8], *(const v8u*)&r[16 + lh * 8]);
    }
#pragma unroll
    for (int ni = 0; ni < 2; ++ni) { // B frag: col = lane%16, K chunks by lane/16
      const u16* r = &Bs[wn * 32 + ni * 16 + lm][0];
      b[ni] = cat16(*(const v8u*)&r[lh * 8], *(const v8u*)&r[16 + lh * 8]);
    }
#pragma unroll
    for (int mi = 0; mi < 2; ++mi)
#pragma unroll
      for (int ni = 0; ni < 2; ++ni)
        c[mi][ni] = wmma_bf16(a[mi], b[ni], c[mi][ni]);
    __syncthreads();
  }

#pragma unroll
  for (int mi = 0; mi < 2; ++mi)
#pragma unroll
    for (int ni = 0; ni < 2; ++ni)
#pragma unroll
      for (int r = 0; r < 8; ++r) {
        int row = m0 + wm * 32 + mi * 16 + r + 8 * lh;  // C layout
        int col = n0 + wn * 32 + ni * 16 + lm;
        float v = c[mi][ni][r];
        if constexpr (sizeof(TOUT) == 2)
          ((u16*)C)[(size_t)row * N + col] = f2bf(v);
        else
          ((float*)C)[(size_t)row * N + col] = v;
      }
}

// ---------------------------------------------------------------------------
// RoPE + relayout [b,s,h,dk] -> [b,h,s,dk]; one thread per (even,odd) pair.
// `scale` folds 1/sqrt(dk) into Q (2^-3, exact in bf16).
// ---------------------------------------------------------------------------
__global__ void rope_kernel(const u16* __restrict__ in,
                            u16* __restrict__ out, int n, float scale) {
  int i = blockIdx.x * blockDim.x + threadIdx.x;  // ((b*2048+t)*16+h)*32+j
  if (i >= n) return;
  int j    = i & 31;
  int h    = (i >> 5) & 15;
  int tpos = (i >> 9) & 2047;
  int b    = i >> 20;
  float fr  = __expf(-9.210340371976184f * ((float)j * (1.0f / 32.0f)));
  float ang = (float)tpos * fr;
  float cs = cosf(ang), sn = sinf(ang);
  size_t src = (size_t)i * 2;
  float xe = bf2f(in[src]), xo = bf2f(in[src + 1]);
  size_t dst = ((size_t)((b << 4) + h) * 2048 + tpos) * 64 + (j << 1);
  out[dst]     = f2bf((xe * cs - xo * sn) * scale);
  out[dst + 1] = f2bf((xe * sn + xo * cs) * scale);
}

// ---------------------------------------------------------------------------
// V relayout [b,s,h,dk] -> Vt [b,h,dk,s]  (k-contiguous B operand for P*V)
// ---------------------------------------------------------------------------
__global__ void vtrans_kernel(const u16* __restrict__ in,
                              u16* __restrict__ out) {
  int i = blockIdx.x * blockDim.x + threadIdx.x;  // ((b*2048+t)*16+h)*64+d
  int d = i & 63, h = (i >> 6) & 15, tpos = (i >> 10) & 2047, b = i >> 21;
  out[((size_t)((b << 4) + h) * 64 + d) * 2048 + tpos] = in[i];
}

// ---------------------------------------------------------------------------
// Flash attention: one wave per 16-query tile, online softmax over 32-key
// blocks. 8 WMMAs per block (4 for Q*K^T, 4 for P*V).  Q pre-scaled.
// Q,K: [b,h,s,64] bf16   Vt: [b,h,64,s] bf16   O: [b,s,h*64] bf16
// ---------------------------------------------------------------------------
__global__ void flash_kernel(const u16* __restrict__ Q,
                             const u16* __restrict__ Kh,
                             const u16* __restrict__ Vt,
                             u16* __restrict__ O) {
  __shared__ u16 P[8][512];  // per-wave 16x32 prob tile (C-layout -> A-layout)
  const int lane = threadIdx.x & 31, wid = threadIdx.x >> 5;
  const int lm = lane & 15, lh = lane >> 4;
  const int h = blockIdx.y, b = blockIdx.z;
  const int bh = b * 16 + h;
  const int qb = (blockIdx.x * 8 + wid) * 16;

  // Q A-frags for dk 0..31 and 32..63 (kept in registers for all key blocks)
  const u16* qrow = Q + ((size_t)bh * 2048 + qb + lm) * 64;
  v16u qa0 = cat16(*(const v8u*)(qrow + lh * 8),      *(const v8u*)(qrow + 16 + lh * 8));
  v16u qa1 = cat16(*(const v8u*)(qrow + 32 + lh * 8), *(const v8u*)(qrow + 48 + lh * 8));

  v8f o[4] = {};
  float mrow[8], lrow[8];
#pragma unroll
  for (int r = 0; r < 8; ++r) { mrow[r] = -3.0e38f; lrow[r] = 0.0f; }

  const int nkb = ((qb + 15) >> 5) + 1;
  for (int kb = 0; kb < nkb; ++kb) {
    const int k0 = kb * 32;
    if (kb + 1 < nkb) {  // prefetch next key block (global_prefetch_b8)
      __builtin_prefetch(Kh + ((size_t)bh * 2048 + k0 + 32 + lm * 2) * 64, 0, 1);
      __builtin_prefetch(Vt + ((size_t)bh * 64 + lane) * 2048 + k0 + 32, 0, 1);
    }
    v8f s[2] = {};
#pragma unroll
    for (int st = 0; st < 2; ++st) {  // two 16-key tiles
      const u16* krow = Kh + ((size_t)bh * 2048 + k0 + st * 16 + lm) * 64;
      v16u b0 = cat16(*(const v8u*)(krow + lh * 8),      *(const v8u*)(krow + 16 + lh * 8));
      v16u b1 = cat16(*(const v8u*)(krow + 32 + lh * 8), *(const v8u*)(krow + 48 + lh * 8));
      s[st] = wmma_bf16(qa0, b0, s[st]);
      s[st] = wmma_bf16(qa1, b1, s[st]);
    }
    // causal mask + online softmax (C rows live in 16-lane halves; all
    // row-wise reductions are DPP16 butterflies in the VALU)
#pragma unroll
    for (int r = 0; r < 8; ++r) {
      const int row = qb + r + 8 * lh;
      float s0 = s[0][r];
      float s1 = s[1][r];
      if (k0 + lm      > row) s0 = -3.0e38f;
      if (k0 + 16 + lm > row) s1 = -3.0e38f;
      float mnew = fmaxf(mrow[r], redmax16(fmaxf(s0, s1)));
      float p0 = __expf(s0 - mnew);
      float p1 = __expf(s1 - mnew);
      float rs = redsum16(p0 + p1);
      float fac = __expf(mrow[r] - mnew);
      mrow[r] = mnew;
      lrow[r] = lrow[r] * fac + rs;
#pragma unroll
      for (int nt = 0; nt < 4; ++nt) o[nt][r] *= fac;
      P[wid][(r + 8 * lh) * 32 + lm]      = f2bf(p0);
      P[wid][(r + 8 * lh) * 32 + 16 + lm] = f2bf(p1);
    }
    // P: C-layout -> A-layout through per-wave LDS (DS ops are wave-ordered)
    const u16* prow = &P[wid][lm * 32];
    v16u pa = cat16(*(const v8u*)(prow + lh * 8), *(const v8u*)(prow + 16 + lh * 8));
#pragma unroll
    for (int nt = 0; nt < 4; ++nt) {
      const u16* vrow = Vt + ((size_t)bh * 64 + nt * 16 + lm) * 2048 + k0;
      v16u bv = cat16(*(const v8u*)(vrow + lh * 8), *(const v8u*)(vrow + 16 + lh * 8));
      o[nt] = wmma_bf16(pa, bv, o[nt]);
    }
  }
  // normalize and store attn in [b,s,h*64] so the output GEMM is K-contiguous
#pragma unroll
  for (int r = 0; r < 8; ++r) {
    float inv = 1.0f / lrow[r];
    int row = qb + r + 8 * lh;
#pragma unroll
    for (int nt = 0; nt < 4; ++nt)
      O[((size_t)b * 2048 + row) * 1024 + h * 64 + nt * 16 + lm] =
          f2bf(o[nt][r] * inv);
  }
}

// ---------------------------------------------------------------------------
extern "C" void kernel_launch(void* const* d_in, const int* in_sizes, int n_in,
                              void* d_out, int out_size, void* d_ws,
                              size_t ws_size, hipStream_t stream) {
  const float* x  = (const float*)d_in[0];
  const float* Wq = (const float*)d_in[1];
  const float* Wk = (const float*)d_in[2];
  const float* Wv = (const float*)d_in[3];
  const float* Wo = (const float*)d_in[4];

  const size_t BS = 4 * 2048;     // 8192 rows
  const size_t D  = 1024;
  const size_t NX = BS * D;       // 8,388,608 elems
  const size_t NW = D * D;

  char* p = (char*)d_ws;
  auto carve = [&](size_t elems) -> u16* {
    u16* r = (u16*)p;
    p += ((elems * 2 + 255) & ~(size_t)255);
    return r;
  };
  u16* xb = carve(NX);
  u16* wq = carve(NW); u16* wk = carve(NW); u16* wv = carve(NW); u16* wo = carve(NW);
  u16* qp = carve(NX); u16* kp = carve(NX); u16* vp = carve(NX);
  u16* qh = carve(NX); u16* kh = carve(NX); u16* vt = carve(NX);
  u16* at = carve(NX);

  const int TPB = 256;
  cvt_bf16_kernel<<<(NX + TPB - 1) / TPB, TPB, 0, stream>>>(x, xb, (int)NX);
  cvt_bf16_kernel<<<(NW + TPB - 1) / TPB, TPB, 0, stream>>>(Wq, wq, (int)NW);
  cvt_bf16_kernel<<<(NW + TPB - 1) / TPB, TPB, 0, stream>>>(Wk, wk, (int)NW);
  cvt_bf16_kernel<<<(NW + TPB - 1) / TPB, TPB, 0, stream>>>(Wv, wv, (int)NW);
  cvt_bf16_kernel<<<(NW + TPB - 1) / TPB, TPB, 0, stream>>>(Wo, wo, (int)NW);

  dim3 gg(1024 / 64, 8192 / 128);  // (16, 64)
  gemm_abt_kernel<u16><<<gg, 256, 0, stream>>>(xb, wq, qp, 8192, 1024, 1024);
  gemm_abt_kernel<u16><<<gg, 256, 0, stream>>>(xb, wk, kp, 8192, 1024, 1024);
  gemm_abt_kernel<u16><<<gg, 256, 0, stream>>>(xb, wv, vp, 8192, 1024, 1024);

  const int npair = 4 * 2048 * 16 * 32;
  rope_kernel<<<(npair + TPB - 1) / TPB, TPB, 0, stream>>>(qp, qh, npair, 0.125f);
  rope_kernel<<<(npair + TPB - 1) / TPB, TPB, 0, stream>>>(kp, kh, npair, 1.0f);
  vtrans_kernel<<<(NX + TPB - 1) / TPB, TPB, 0, stream>>>(vp, vt);

  flash_kernel<<<dim3(16, 16, 4), 256, 0, stream>>>(qh, kh, vt, at);

  gemm_abt_kernel<float><<<gg, 256, 0, stream>>>(at, wo, (float*)d_out,
                                                 8192, 1024, 1024);
}
